// GIN_66185446031495
// MI455X (gfx1250) — matrix operand order repaired and checked by reading, not compiled, not run on hardware
//
#include <hip/hip_runtime.h>
#include <hip/hip_bf16.h>
#include <cstddef>
#include <cstdint>

// ---------------------------------------------------------------------------
// GIN forward on MI455X (gfx1250):
//  - bf16 WMMA GEMMs (fp32 accum), B pre-packed in fragment order (L2 resident)
//  - A tiles staged via global_load_async_to_lds_b128 (ASYNCcnt), double buffered
//  - fp32 atomic scatter aggregation into L2-resident agg buffer
// ---------------------------------------------------------------------------

typedef __attribute__((ext_vector_type(16))) __bf16 v16bf;
typedef __attribute__((ext_vector_type(8)))  float  v8f;

namespace gin {
constexpr int N_NODES = 50000;
constexpr int N_EDGES = 625000;
constexpr int IN_DIM  = 512;
constexpr int HID     = 128;
constexpr int OUT_DIM = 40;
constexpr int NL      = 3;

constexpr int ASTF = 36;   // fp32 LDS row stride for A tile (16x32 used) — kills bank conflicts
constexpr int NTP  = 8;    // B packed to 8 n-tiles (128 cols), zero padded
}

// ------------------------- CDNA5 async copy helpers -------------------------

__device__ __forceinline__ void async_lds_b128(unsigned lds_byte_off,
                                               unsigned g_byte_off,
                                               const void* sbase) {
    // GLOBAL_LOAD_ASYNC_TO_LDS_B128 (GVS mode): copies 16 bytes at saddr + vaddr to LDS byte address vdst
    asm volatile("global_load_async_to_lds_b128 %0, %1, %2"
                 :: "v"(lds_byte_off), "v"(g_byte_off), "s"(sbase)
                 : "memory");
}
__device__ __forceinline__ void wait_async0() {
    asm volatile("s_wait_asynccnt 0x0" ::: "memory");
}
__device__ __forceinline__ unsigned lds_off_u32(const void* p) {
    // low 32 bits of a generic LDS pointer == wave-relative LDS byte address
    return (unsigned)(uintptr_t)p;
}

// ---------------------------- helper kernels -------------------------------

__global__ void fill_zero_f4(float4* __restrict__ p, int n4) {
    int i = blockIdx.x * blockDim.x + threadIdx.x;
    if (i < n4) p[i] = make_float4(0.f, 0.f, 0.f, 0.f);
}

// Pack fp32 weight matrix W[K][N] (row-major) into bf16 WMMA B-fragment order,
// zero-padded to NTP n-tiles:
//   flat = ((nt*KT + ks)*32 + lane)*16 + i
//   col  = nt*16 + (lane&15),  k = ks*32 + (lane<16 ? 0 : 8) + (i&7) + ((i>>3)<<4)
__global__ void pack_b_bf16(const float* __restrict__ W, __bf16* __restrict__ Bp,
                            int K, int N) {
    const int KT = K >> 5;
    const int total = gin::NTP * KT * 512;
    int idx = blockIdx.x * blockDim.x + threadIdx.x;
    if (idx >= total) return;
    const int i    = idx & 15;
    const int lane = (idx >> 4) & 31;
    const int rest = idx >> 9;          // nt*KT + ks
    const int ks   = rest % KT;
    const int nt   = rest / KT;
    const int col  = nt * 16 + (lane & 15);
    const int k    = ks * 32 + ((lane < 16) ? 0 : 8) + (i & 7) + ((i >> 3) << 4);
    const float v  = (col < N) ? W[(size_t)k * N + col] : 0.f;
    Bp[idx] = (__bf16)v;
}

// One wave per edge: 32 lanes x float4 = 128 channels.
// agg[dst[e], :] += edge_w[e] * h[src[e], :]
__global__ __launch_bounds__(256) void scatter_edges(
    const float* __restrict__ h, const int* __restrict__ src,
    const int* __restrict__ dst, const float* __restrict__ ew,
    float* __restrict__ agg, int n_edges)
{
    const int lane = threadIdx.x & 31;
    const int e = blockIdx.x * (blockDim.x >> 5) + (threadIdx.x >> 5);
    if (e >= n_edges) return;
    const int   s = src[e];
    const int   d = dst[e];
    const float w = ew[e];
    const float4 v = *(const float4*)(h + (size_t)s * gin::HID + lane * 4);
    float* a = agg + (size_t)d * gin::HID + lane * 4;
    atomicAdd(a + 0, v.x * w);
    atomicAdd(a + 1, v.y * w);
    atomicAdd(a + 2, v.z * w);
    atomicAdd(a + 3, v.w * w);
}

// ------------------------------ WMMA GEMM ----------------------------------
// C[M,N] = act( (A (+A2)) @ B + bias ),  A,A2,bias,C fp32; Bp packed bf16.
// Block: 128 threads (4 waves). Block tile 16x128: wave w owns n-tiles
// {2w, 2w+1} (two accumulators, one shared A fragment). K step 32.
// A staged fp32 via async-to-LDS, double buffered; converted to bf16 at
// fragment-assembly time (fuses the h+agg skip add when HAS_A2).
template <bool HAS_A2>
__global__ __launch_bounds__(128) void wmma_gemm(
    const float* __restrict__ A, const float* __restrict__ A2,
    const __bf16* __restrict__ Bp, const float* __restrict__ bias,
    float* __restrict__ C, int N, int K, int do_relu)
{
    using namespace gin;
    __shared__ alignas(16) float Asm [2][16 * ASTF];
    __shared__ alignas(16) float A2sm[HAS_A2 ? 2 : 1][HAS_A2 ? 16 * ASTF : 1];

    const int tid   = threadIdx.x;
    const int lane  = tid & 31;
    const int wave  = tid >> 5;
    const int mbase = blockIdx.x * 16;
    const int KT    = K >> 5;

    // this thread's async-copy slot: 4 consecutive floats of the 16x32 tile
    const int ar = tid >> 3;            // row 0..15
    const int ac = (tid & 7) * 4;       // col 0,4,...,28
    const unsigned ldsSlotA[2] = {
        lds_off_u32(&Asm[0][ar * ASTF + ac]),
        lds_off_u32(&Asm[1][ar * ASTF + ac]) };
    unsigned ldsSlotA2[2] = {0, 0};
    if constexpr (HAS_A2) {
        ldsSlotA2[0] = lds_off_u32(&A2sm[0][ar * ASTF + ac]);
        ldsSlotA2[1] = lds_off_u32(&A2sm[1][ar * ASTF + ac]);
    }
    const unsigned gRowOff = (unsigned)(((size_t)(mbase + ar) * K + ac) * sizeof(float));

    // fragment geometry (CDNA5 WMMA 16-bit A layout, CDNA5 ISA)
    const int am   = lane & 15;
    const int akb0 = (lane < 16) ? 0 : 8;

    // B fragment stream: ((nt*KT + ks)*32 + lane)*16 elements
    const int nt0 = wave * 2;
    const __bf16* bp0 = Bp + ((size_t)(nt0 * KT) * 32 + lane) * 16;
    const __bf16* bp1 = Bp + ((size_t)((nt0 + 1) * KT) * 32 + lane) * 16;

    v8f acc0 = {}, acc1 = {};

    // prologue: stage tile 0 into buffer 0
    async_lds_b128(ldsSlotA[0], gRowOff, A);
    if constexpr (HAS_A2) async_lds_b128(ldsSlotA2[0], gRowOff, A2);
    wait_async0();
    __syncthreads();

    for (int ks = 0; ks < KT; ++ks) {
        const int cur = ks & 1;
        if (ks + 1 < KT) {                 // prefetch next tile into other buffer
            const unsigned go = gRowOff + (unsigned)((ks + 1) * 32 * sizeof(float));
            async_lds_b128(ldsSlotA[cur ^ 1], go, A);
            if constexpr (HAS_A2) async_lds_b128(ldsSlotA2[cur ^ 1], go, A2);
        }

        // --- assemble bf16 A fragment from fp32 LDS tile ---
        const float* arow = &Asm[cur][am * ASTF];
        float4 x0 = *(const float4*)(arow + akb0);
        float4 x1 = *(const float4*)(arow + akb0 + 4);
        float4 x2 = *(const float4*)(arow + 16 + akb0);
        float4 x3 = *(const float4*)(arow + 16 + akb0 + 4);
        if constexpr (HAS_A2) {
            const float* brow = &A2sm[cur][am * ASTF];
            const float4 y0 = *(const float4*)(brow + akb0);
            const float4 y1 = *(const float4*)(brow + akb0 + 4);
            const float4 y2 = *(const float4*)(brow + 16 + akb0);
            const float4 y3 = *(const float4*)(brow + 16 + akb0 + 4);
            x0.x += y0.x; x0.y += y0.y; x0.z += y0.z; x0.w += y0.w;
            x1.x += y1.x; x1.y += y1.y; x1.z += y1.z; x1.w += y1.w;
            x2.x += y2.x; x2.y += y2.y; x2.z += y2.z; x2.w += y2.w;
            x3.x += y3.x; x3.y += y3.y; x3.z += y3.z; x3.w += y3.w;
        }
        union { v16bf v; __bf16 e[16]; } af;
        const float f[16] = { x0.x, x0.y, x0.z, x0.w,  x1.x, x1.y, x1.z, x1.w,
                              x2.x, x2.y, x2.z, x2.w,  x3.x, x3.y, x3.z, x3.w };
        #pragma unroll
        for (int i = 0; i < 16; ++i) af.e[i] = (__bf16)f[i];

        // --- B fragments: straight 32B global loads from packed stream ---
        const v16bf b0 = *(const v16bf*)(bp0 + (size_t)ks * 512);
        const v16bf b1 = *(const v16bf*)(bp1 + (size_t)ks * 512);

        acc0 = __builtin_amdgcn_wmma_f32_16x16x32_bf16(
            false, af.v, false, b0, (short)0, acc0, false, false);
        acc1 = __builtin_amdgcn_wmma_f32_16x16x32_bf16(
            false, af.v, false, b1, (short)0, acc1, false, false);

        wait_async0();          // next tile landed (no-op on last iteration)
        __syncthreads();        // all waves' copies visible / reads done
    }

    // --- epilogue: C/D layout -> row = r + 8*(lane>=16), col = lane&15 ---
    const int rbase = mbase + ((lane >> 4) << 3);
    const int c0 = nt0 * 16 + (lane & 15);
    const int c1 = c0 + 16;
    if (c0 < N) {
        const float bv = bias ? bias[c0] : 0.f;
        #pragma unroll
        for (int r = 0; r < 8; ++r) {
            float v = acc0[r] + bv;
            if (do_relu) v = fmaxf(v, 0.f);
            C[(size_t)(rbase + r) * N + c0] = v;
        }
    }
    if (c1 < N) {
        const float bv = bias ? bias[c1] : 0.f;
        #pragma unroll
        for (int r = 0; r < 8; ++r) {
            float v = acc1[r] + bv;
            if (do_relu) v = fmaxf(v, 0.f);
            C[(size_t)(rbase + r) * N + c1] = v;
        }
    }
}

// ------------------------------- launcher ----------------------------------

extern "C" void kernel_launch(void* const* d_in, const int* in_sizes, int n_in,
                              void* d_out, int out_size, void* d_ws, size_t ws_size,
                              hipStream_t stream) {
    using namespace gin;
    const float* features = (const float*)d_in[0];
    const int*   src      = (const int*)  d_in[1];
    const int*   dst      = (const int*)  d_in[2];
    const float* edge_w   = (const float*)d_in[3];
    const float* emb_W    = (const float*)d_in[4];
    const float* emb_b    = (const float*)d_in[5];
    const float* W1       = (const float*)d_in[6];
    const float* b1       = (const float*)d_in[7];
    const float* W2       = (const float*)d_in[8];
    const float* b2       = (const float*)d_in[9];
    const float* ro_W     = (const float*)d_in[10];
    const float* ro_b     = (const float*)d_in[11];
    float* out = (float*)d_out;

    // Workspace layout
    char* ws = (char*)d_ws;
    const size_t act_elems = (size_t)N_NODES * HID;             // 6.4M floats
    float*  h    = (float*)ws;   ws += act_elems * sizeof(float);
    float*  agg  = (float*)ws;   ws += act_elems * sizeof(float);
    float*  t    = (float*)ws;   ws += act_elems * sizeof(float);
    const size_t bpE = (size_t)NTP * (IN_DIM / 32) * 512;       // 65536
    const size_t bpH = (size_t)NTP * (HID   / 32) * 512;        // 16384
    __bf16* wbE = (__bf16*)ws;   ws += bpE * sizeof(__bf16);
    __bf16* wb1 = (__bf16*)ws;   ws += (size_t)NL * bpH * sizeof(__bf16);
    __bf16* wb2 = (__bf16*)ws;   ws += (size_t)NL * bpH * sizeof(__bf16);
    __bf16* wbR = (__bf16*)ws;   ws += bpH * sizeof(__bf16);

    // 1) pack weights (fp32 -> bf16, fragment order, zero-padded to 128 cols)
    pack_b_bf16<<<(int)((bpE + 255) / 256), 256, 0, stream>>>(emb_W, wbE, IN_DIM, HID);
    for (int l = 0; l < NL; ++l) {
        pack_b_bf16<<<(int)((bpH + 255) / 256), 256, 0, stream>>>(
            W1 + (size_t)l * HID * HID, wb1 + (size_t)l * bpH, HID, HID);
        pack_b_bf16<<<(int)((bpH + 255) / 256), 256, 0, stream>>>(
            W2 + (size_t)l * HID * HID, wb2 + (size_t)l * bpH, HID, HID);
    }
    pack_b_bf16<<<(int)((bpH + 255) / 256), 256, 0, stream>>>(ro_W, wbR, HID, OUT_DIM);

    const dim3 blk(128);
    const dim3 grid(N_NODES / 16);                  // 3125 blocks, tile 16x128

    // 2) embedding: h = features @ emb_W + emb_b
    wmma_gemm<false><<<grid, blk, 0, stream>>>(
        features, nullptr, wbE, emb_b, h, HID, IN_DIM, /*relu=*/0);

    // 3) GIN layers
    const int n4 = (int)(act_elems / 4);
    const int edge_blocks = (N_EDGES + 7) / 8;      // 8 waves per 256-thr block
    for (int l = 0; l < NL; ++l) {
        fill_zero_f4<<<(n4 + 255) / 256, 256, 0, stream>>>((float4*)agg, n4);
        scatter_edges<<<edge_blocks, 256, 0, stream>>>(h, src, dst, edge_w, agg, N_EDGES);
        // t = relu((h + agg) @ W1[l] + b1[l])
        wmma_gemm<true><<<grid, blk, 0, stream>>>(
            h, agg, wb1 + (size_t)l * bpH, b1 + (size_t)l * HID,
            t, HID, HID, /*relu=*/1);
        // h = relu(t @ W2[l] + b2[l])
        wmma_gemm<false><<<grid, blk, 0, stream>>>(
            t, nullptr, wb2 + (size_t)l * bpH, b2 + (size_t)l * HID,
            h, HID, HID, /*relu=*/1);
    }

    // 4) readout: out = h @ ro_W + ro_b   (N = 40; padded tiles, guarded stores)
    wmma_gemm<false><<<grid, blk, 0, stream>>>(
        h, nullptr, wbR, ro_b, out, OUT_DIM, HID, /*relu=*/0);
}
